// MSA_31078383354471
// MI455X (gfx1250) — matrix operand order repaired and checked by reading, not compiled, not run on hardware
//
#include <hip/hip_runtime.h>
#include <hip/hip_bf16.h>
#include <math.h>

typedef __attribute__((ext_vector_type(16))) __bf16 v16bf;
typedef __attribute__((ext_vector_type(8)))  __bf16 v8bf;
typedef __attribute__((ext_vector_type(8)))  float  v8f;

#define B_    16
#define C_    384
#define N_    1024
#define NH_   12
#define DH_   32
// (1/sqrt(32)) * log2(e): softmax computed in base-2 (v_exp_f32 is exp2)
#define QK_SCALE_ 0.2550565445441432f

static __device__ __forceinline__ v8f wmma_bf16(const v16bf& a, const v16bf& b, const v8f& c) {
  return __builtin_amdgcn_wmma_f32_16x16x32_bf16(false, a, false, b, (short)0, c, false, false);
}

// 32-bit LDS offset for async-to-LDS (low 32 bits of the LDS aperture address)
static __device__ __forceinline__ unsigned lds_off(const void* p) {
  return (unsigned)(size_t)p;
}
// GLOBAL_LOAD_ASYNC_TO_LDS_B128: per-lane global addr -> per-lane LDS addr (ASYNCcnt)
static __device__ __forceinline__ void async_copy_b128(unsigned loff, const void* gsrc) {
  asm volatile("global_load_async_to_lds_b128 %0, %1, off"
               :: "v"(loff), "v"(gsrc) : "memory");
}
static __device__ __forceinline__ void async_wait0() {
  asm volatile("s_wait_asynccnt 0x0" ::: "memory");
}
// Partial wait: allow the newest 2 async ops (next tile) to stay in flight;
// in-order completion => older tile's ops are done.
static __device__ __forceinline__ void async_wait2() {
  asm volatile("s_wait_asynccnt 0x2" ::: "memory");
}

// Fragment-layout LDS staging (per ISA 7.12.2):
//  A-frag (16x32, MxK): lane = row + 16*((k>>3)&1), elem = (k&7) + (k>=16 ? 8 : 0)
//  B-frag (32x16, KxN): lane = col + 16*((k>>3)&1), elem = (k&7) + (k>=16 ? 8 : 0)

// ---------------- Kernel 1: QKV projection (fp32 in -> bf16 q/k/v) ----------------
__global__ __launch_bounds__(128) void qkv_kernel(const float* __restrict__ x,
                                                  const float* __restrict__ qkv_w,
                                                  __bf16* __restrict__ qws,
                                                  __bf16* __restrict__ kws,
                                                  __bf16* __restrict__ vws) {
  __shared__ __align__(32) __bf16 AF[4][32][16];
  __shared__ __align__(32) __bf16 BF[4][32][16];

  const int tid  = threadIdx.x;
  const int wave = tid >> 5;
  const int lane = tid & 31;
  const int half = lane >> 4;
  const int l15  = lane & 15;

  const int b  = (blockIdx.x * 64) / N_;
  const int n0 = (blockIdx.x * 64) % N_;
  const int o0 = blockIdx.y * 64;

  const v8f vzero = {0.f,0.f,0.f,0.f,0.f,0.f,0.f,0.f};
  v8f acc[4];
  #pragma unroll
  for (int t = 0; t < 4; ++t) acc[t] = vzero;

  for (int c0 = 0; c0 < C_; c0 += 32) {
    {  // A tile: x[b,k,n] contiguous along n (coalesced scalar loads) + cvt
      const int nn = tid & 63;
      const int wt = nn >> 4, lrow = nn & 15;
      #pragma unroll
      for (int gg = 0; gg < 2; ++gg) {
        const int g = (tid >> 6) + 2 * gg;
        v8bf pk;
        #pragma unroll
        for (int kk = 0; kk < 8; ++kk)
          pk[kk] = (__bf16)x[((size_t)b * C_ + (c0 + g * 8 + kk)) * N_ + n0 + nn];
        *(v8bf*)&AF[wt][lrow + 16 * (g & 1)][(g >> 1) * 8] = pk;
      }
    }
    {  // B tile: qkv_w[o,k] contiguous along k + cvt
      const int g = tid & 3;
      #pragma unroll
      for (int oi = 0; oi < 2; ++oi) {
        const int oo = (tid >> 2) + 32 * oi;
        const float4* src = (const float4*)&qkv_w[(size_t)(o0 + oo) * C_ + c0 + g * 8];
        float4 f0 = src[0], f1 = src[1];
        v8bf pk;
        pk[0]=(__bf16)f0.x; pk[1]=(__bf16)f0.y; pk[2]=(__bf16)f0.z; pk[3]=(__bf16)f0.w;
        pk[4]=(__bf16)f1.x; pk[5]=(__bf16)f1.y; pk[6]=(__bf16)f1.z; pk[7]=(__bf16)f1.w;
        *(v8bf*)&BF[oo >> 4][(oo & 15) + 16 * (g & 1)][(g >> 1) * 8] = pk;
      }
    }
    __syncthreads();

    const v16bf a = *(const v16bf*)&AF[wave][lane][0];
    #pragma unroll
    for (int t = 0; t < 4; ++t) {
      const v16bf bb = *(const v16bf*)&BF[t][lane][0];
      acc[t] = wmma_bf16(a, bb, acc[t]);
    }
    __syncthreads();
  }

  #pragma unroll
  for (int t = 0; t < 4; ++t) {
    #pragma unroll
    for (int r = 0; r < 8; ++r) {
      int n = n0 + wave * 16 + r + (half ? 8 : 0);
      int o = o0 + t * 16 + l15;
      float v = acc[t][r];
      int oc = (o < C_) ? o : (o < 2 * C_ ? o - C_ : o - 2 * C_);
      int hh = oc >> 5, dd = oc & 31;
      size_t idx = (((size_t)b * NH_ + hh) * N_ + n) * DH_ + dd;
      if (o < C_)            qws[idx] = (__bf16)(v * QK_SCALE_);   // scale+log2e folded
      else if (o < 2 * C_)   kws[idx] = (__bf16)v;
      else                   vws[idx] = (__bf16)v;
    }
  }
}

// ---------------- Kernel 2: flash attention, 64-key tiles, double-buffered async ----------------
__global__ __launch_bounds__(128) void attn_kernel(const __bf16* __restrict__ qws,
                                                   const __bf16* __restrict__ kws,
                                                   const __bf16* __restrict__ vws,
                                                   __bf16* __restrict__ ao) {
  __shared__ __align__(32) __bf16 QF[4][32][16];          // q rows, A-frag layout
  __shared__ __align__(32) __bf16 KF[2][4][32][16];       // double-buffered k^T, B-frag
  __shared__ __align__(32) __bf16 VF[2][2][2][32][16];    // double-buffered v, B-frag
  __shared__ __align__(32) __bf16 PF[4][2][32][16];       // per-wave P, 2 key-half A-frags

  const int tid  = threadIdx.x;
  const int wave = tid >> 5;
  const int lane = tid & 31;
  const int half = lane >> 4;
  const int l15  = lane & 15;

  const int bh = blockIdx.x;
  const int b  = bh / NH_, hh = bh % NH_;
  const size_t base = (size_t)bh * N_ * DH_;
  const int q0 = blockIdx.y * 64;

  // stage one 64-key K/V tile into buffer `buf` (K via async DMA, V via VGPR scatter)
  auto stage_kv = [&](int buf, int kb) {
    const int g = tid & 3;
    #pragma unroll
    for (int ki = 0; ki < 2; ++ki) {
      const int kk = (tid >> 2) + 32 * ki;   // 0..63
      async_copy_b128(lds_off(&KF[buf][kk >> 4][(kk & 15) + 16 * (g & 1)][(g >> 1) * 8]),
                      &kws[base + (size_t)(kb + kk) * DH_ + g * 8]);
      v8bf vd = *(const v8bf*)&vws[base + (size_t)(kb + kk) * DH_ + g * 8];
      const int kh = kk >> 5, kl = kk & 31;
      const int lbase = (g & 1) * 8 + (((kl >> 3) & 1) ? 16 : 0);
      const int dt = g >> 1, vi = (kl & 7) + ((kl >= 16) ? 8 : 0);
      #pragma unroll
      for (int dd = 0; dd < 8; ++dd)
        VF[buf][kh][dt][lbase + dd][vi] = vd[dd];
    }
  };

  {  // Q: pure bf16 b128 copy -> async to LDS in A-frag layout
    const int g = tid & 3;
    #pragma unroll
    for (int ri = 0; ri < 2; ++ri) {
      const int row = (tid >> 2) + 32 * ri;
      async_copy_b128(lds_off(&QF[row >> 4][(row & 15) + 16 * (g & 1)][(g >> 1) * 8]),
                      &qws[base + (size_t)(q0 + row) * DH_ + g * 8]);
    }
  }
  stage_kv(0, 0);          // prologue tile
  async_wait0();
  __syncthreads();
  const v16bf aq = *(const v16bf*)&QF[wave][lane][0];

  const v8f vzero = {0.f,0.f,0.f,0.f,0.f,0.f,0.f,0.f};
  v8f oacc0 = vzero, oacc1 = vzero;
  float m[8], l[8];
  #pragma unroll
  for (int r = 0; r < 8; ++r) { m[r] = -3.0e38f; l[r] = 0.f; }

  for (int it = 0; it < N_ / 64; ++it) {       // 16 iterations
    const int buf = it & 1;
    if (it + 1 < N_ / 64) {
      stage_kv(buf ^ 1, (it + 1) * 64);        // next tile's DMA in flight during compute
      async_wait2();                           // oldest 2 (current tile) complete
    } else {
      async_wait0();
    }
    __syncthreads();

    // S = (q*scale') . k^T for 64 keys (base-2 logits)
    v8f s0 = wmma_bf16(aq, *(const v16bf*)&KF[buf][0][lane][0], vzero);
    v8f s1 = wmma_bf16(aq, *(const v16bf*)&KF[buf][1][lane][0], vzero);
    v8f s2 = wmma_bf16(aq, *(const v16bf*)&KF[buf][2][lane][0], vzero);
    v8f s3 = wmma_bf16(aq, *(const v16bf*)&KF[buf][3][lane][0], vzero);

    const int hk = (l15 >> 3) & 1;
    #pragma unroll
    for (int r = 0; r < 8; ++r) {
      float t = fmaxf(fmaxf(s0[r], s1[r]), fmaxf(s2[r], s3[r]));
      t = fmaxf(t, __shfl_xor(t, 1));
      t = fmaxf(t, __shfl_xor(t, 2));
      t = fmaxf(t, __shfl_xor(t, 4));
      t = fmaxf(t, __shfl_xor(t, 8));
      float mn   = fmaxf(m[r], t);
      float corr = exp2f(m[r] - mn);
      float p0 = exp2f(s0[r] - mn);
      float p1 = exp2f(s1[r] - mn);
      float p2 = exp2f(s2[r] - mn);
      float p3 = exp2f(s3[r] - mn);
      float rs = (p0 + p1) + (p2 + p3);
      rs += __shfl_xor(rs, 1);
      rs += __shfl_xor(rs, 2);
      rs += __shfl_xor(rs, 4);
      rs += __shfl_xor(rs, 8);
      l[r] = l[r] * corr + rs;
      m[r] = mn;
      oacc0[r] *= corr;
      oacc1[r] *= corr;
      const int dlane = (r + (half ? 8 : 0)) + 16 * hk;   // D-layout -> A-frag
      PF[wave][0][dlane][l15 & 7]       = (__bf16)p0;
      PF[wave][0][dlane][(l15 & 7) + 8] = (__bf16)p1;
      PF[wave][1][dlane][l15 & 7]       = (__bf16)p2;
      PF[wave][1][dlane][(l15 & 7) + 8] = (__bf16)p3;
    }

    // O += P(16x64) . V(64x32): two K-steps of 32 keys, two d-subtiles
    const v16bf ap0 = *(const v16bf*)&PF[wave][0][lane][0];
    const v16bf ap1 = *(const v16bf*)&PF[wave][1][lane][0];
    oacc0 = wmma_bf16(ap0, *(const v16bf*)&VF[buf][0][0][lane][0], oacc0);
    oacc1 = wmma_bf16(ap0, *(const v16bf*)&VF[buf][0][1][lane][0], oacc1);
    oacc0 = wmma_bf16(ap1, *(const v16bf*)&VF[buf][1][0][lane][0], oacc0);
    oacc1 = wmma_bf16(ap1, *(const v16bf*)&VF[buf][1][1][lane][0], oacc1);

    __syncthreads();   // all reads of this buffer done before it is restaged
  }

  #pragma unroll
  for (int r = 0; r < 8; ++r) {
    float inv = 1.0f / l[r];
    int n = q0 + wave * 16 + r + (half ? 8 : 0);
    size_t rowbase = ((size_t)b * N_ + n) * C_ + hh * DH_;
    ao[rowbase + l15]      = (__bf16)(oacc0[r] * inv);
    ao[rowbase + 16 + l15] = (__bf16)(oacc1[r] * inv);
  }
}

// ---------------- Kernel 3: output projection + bias + NCHW transpose ----------------
__global__ __launch_bounds__(128) void proj_kernel(const __bf16* __restrict__ ao,
                                                   const float* __restrict__ proj_w,
                                                   const float* __restrict__ proj_b,
                                                   float* __restrict__ out) {
  __shared__ __align__(32) __bf16 AF[2][4][32][16];   // double-buffered (async DMA)
  __shared__ __align__(32) __bf16 BF[2][4][32][16];

  const int tid  = threadIdx.x;
  const int wave = tid >> 5;
  const int lane = tid & 31;
  const int half = lane >> 4;
  const int l15  = lane & 15;

  const int b  = (blockIdx.x * 64) / N_;
  const int n0 = (blockIdx.x * 64) % N_;
  const int o0 = blockIdx.y * 64;

  auto stage_a = [&](int buf, int c0) {
    const int g = tid & 3;
    #pragma unroll
    for (int ni = 0; ni < 2; ++ni) {
      const int nn = (tid >> 2) + 32 * ni;
      async_copy_b128(lds_off(&AF[buf][nn >> 4][(nn & 15) + 16 * (g & 1)][(g >> 1) * 8]),
                      &ao[((size_t)b * N_ + n0 + nn) * C_ + c0 + g * 8]);
    }
  };

  const v8f vzero = {0.f,0.f,0.f,0.f,0.f,0.f,0.f,0.f};
  v8f acc[4];
  #pragma unroll
  for (int t = 0; t < 4; ++t) acc[t] = vzero;

  stage_a(0, 0);   // prologue A tile

  for (int it = 0; it < C_ / 32; ++it) {       // 12 iterations
    const int c0 = it * 32;
    const int buf = it & 1;
    if (it + 1 < C_ / 32) stage_a(buf ^ 1, c0 + 32);   // next A tile DMA in flight

    {  // B tile (current): proj_w f32 + cvt (VGPR path)
      const int g = tid & 3;
      #pragma unroll
      for (int oi = 0; oi < 2; ++oi) {
        const int oo = (tid >> 2) + 32 * oi;
        const float4* src = (const float4*)&proj_w[(size_t)(o0 + oo) * C_ + c0 + g * 8];
        float4 f0 = src[0], f1 = src[1];
        v8bf pk;
        pk[0]=(__bf16)f0.x; pk[1]=(__bf16)f0.y; pk[2]=(__bf16)f0.z; pk[3]=(__bf16)f0.w;
        pk[4]=(__bf16)f1.x; pk[5]=(__bf16)f1.y; pk[6]=(__bf16)f1.z; pk[7]=(__bf16)f1.w;
        *(v8bf*)&BF[buf][oo >> 4][(oo & 15) + 16 * (g & 1)][(g >> 1) * 8] = pk;
      }
    }
    if (it + 1 < C_ / 32) async_wait2();
    else                  async_wait0();
    __syncthreads();

    const v16bf a = *(const v16bf*)&AF[buf][wave][lane][0];
    #pragma unroll
    for (int t = 0; t < 4; ++t) {
      const v16bf bb = *(const v16bf*)&BF[buf][t][lane][0];
      acc[t] = wmma_bf16(a, bb, acc[t]);
    }
    __syncthreads();
  }

  #pragma unroll
  for (int t = 0; t < 4; ++t) {
    #pragma unroll
    for (int r = 0; r < 8; ++r) {
      int n = n0 + wave * 16 + r + (half ? 8 : 0);
      int o = o0 + t * 16 + l15;
      out[((size_t)b * C_ + o) * N_ + n] = acc[t][r] + proj_b[o];
    }
  }
}

extern "C" void kernel_launch(void* const* d_in, const int* in_sizes, int n_in,
                              void* d_out, int out_size, void* d_ws, size_t ws_size,
                              hipStream_t stream) {
  const float* x      = (const float*)d_in[0];
  const float* qkv_w  = (const float*)d_in[1];
  const float* proj_w = (const float*)d_in[2];
  const float* proj_b = (const float*)d_in[3];
  float* out = (float*)d_out;

  const size_t QSZ = (size_t)B_ * NH_ * N_ * DH_;
  __bf16* qws = (__bf16*)d_ws;
  __bf16* kws = qws + QSZ;
  __bf16* vws = kws + QSZ;
  __bf16* ao  = vws + QSZ;

  qkv_kernel <<<dim3(256, 18), 128, 0, stream>>>(x, qkv_w, qws, kws, vws);
  attn_kernel<<<dim3(192, 16), 128, 0, stream>>>(qws, kws, vws, ao);
  proj_kernel<<<dim3(256, 6),  128, 0, stream>>>(ao, proj_w, proj_b, out);
}
